// WindowedMultiHeadAttention_75685913690429
// MI455X (gfx1250) — compile-verified, matrix operand order checked by reading, not run on hardware
//
#include <hip/hip_runtime.h>

// ---------------------------------------------------------------------------
// Types
// ---------------------------------------------------------------------------
typedef __bf16          bf16x16 __attribute__((ext_vector_type(16)));
typedef float           f32x8   __attribute__((ext_vector_type(8)));
typedef float           f32x4   __attribute__((ext_vector_type(4)));
typedef unsigned int    u32x4   __attribute__((ext_vector_type(4)));
typedef unsigned short  u16x4   __attribute__((ext_vector_type(4)));
typedef int             i32x4   __attribute__((ext_vector_type(4)));
typedef int             i32x8   __attribute__((ext_vector_type(8)));

#define NTOK   131072    // 32*64*64 tokens
#define CDIM   512
#define QKVC   1536
#define NHEAD  16
#define HDIM   32
#define EWIN   64        // tokens per 8x8 window

// A/B fragment: 16 bf16 (32 bytes) per lane; C/D fragment: 8 f32 per lane.
union AFrag { bf16x16 v; u32x4 q[2]; };
union CFrag { f32x8  v;  float f[8]; };

__device__ __forceinline__ unsigned short f2bf(float x) {
  union { float f; unsigned u; } c; c.f = x;
  unsigned r = c.u + 0x7FFFu + ((c.u >> 16) & 1u);   // round-to-nearest-even
  return (unsigned short)(r >> 16);
}

// ---------------------------------------------------------------------------
// Tensor Data Mover: 2D tile load (tile_d0 x tile_d1, bf16) global -> LDS.
// Descriptor packing per CDNA5 D# spec (group0 + group1; groups 2/3 zero).
// ---------------------------------------------------------------------------
__device__ __forceinline__ void tdm_load_2d_bf16(
    unsigned lds_addr, const unsigned short* gptr,
    unsigned tensor_d0, unsigned tensor_d1, unsigned stride0,
    unsigned tile_d0, unsigned tile_d1)
{
  unsigned long long ga = (unsigned long long)gptr;
  u32x4 g0;
  g0[0] = 1u;                                           // count=1, user desc
  g0[1] = lds_addr;                                     // LDS byte address
  g0[2] = (unsigned)(ga & 0xFFFFFFFFu);                 // global_addr[31:0]
  g0[3] = (unsigned)((ga >> 32) & 0x01FFFFFFu) | (2u << 30);  // [56:32] | type=2
  i32x8 g1;
  g1[0] = (int)(1u << 16);                              // data_size = 2 bytes
  g1[1] = (int)((tensor_d0 & 0xFFFFu) << 16);           // tensor_dim0[15:0]
  g1[2] = (int)((tensor_d0 >> 16) | ((tensor_d1 & 0xFFFFu) << 16));
  g1[3] = (int)(((tensor_d1 >> 16) & 0xFFFFu) | (tile_d0 << 16));
  g1[4] = (int)(tile_d1 & 0xFFFFu);                     // tile_dim1 (tile_dim2=0)
  g1[5] = (int)stride0;                                 // tensor_dim0_stride[31:0]
  g1[6] = 0;
  g1[7] = 0;
  i32x4 z4 = {0, 0, 0, 0};
#if defined(__clang_major__) && (__clang_major__ >= 23)
  i32x8 z8 = {0, 0, 0, 0, 0, 0, 0, 0};
  __builtin_amdgcn_tensor_load_to_lds(g0, g1, z4, z4, z8, 0);
#else
  __builtin_amdgcn_tensor_load_to_lds(g0, g1, z4, z4, 0);
#endif
}

__device__ __forceinline__ unsigned lds_off(const void* p) {
  // Generic pointers to LDS carry the LDS byte offset in the low 32 bits.
  return (unsigned)(unsigned long long)p;
}

// ---------------------------------------------------------------------------
// Kernel 1: fp32 -> bf16 bulk convert (vectorized 4/thread)
// ---------------------------------------------------------------------------
__global__ __launch_bounds__(256) void cvt_bf16_kernel(
    const float* __restrict__ in, unsigned short* __restrict__ out, long long n4)
{
  long long i = (long long)blockIdx.x * blockDim.x + threadIdx.x;
  if (i < n4) {
    f32x4 f = *(const f32x4*)(in + i * 4);
    u16x4 o;
    o[0] = f2bf(f[0]); o[1] = f2bf(f[1]); o[2] = f2bf(f[2]); o[3] = f2bf(f[3]);
    *(u16x4*)(out + i * 4) = o;
  }
}

// ---------------------------------------------------------------------------
// Kernel 2: W[K][N] fp32 -> Wt[N][K] bf16 (weights are tiny; simplicity wins)
// ---------------------------------------------------------------------------
__global__ __launch_bounds__(256) void transpose_bf16_kernel(
    const float* __restrict__ w, unsigned short* __restrict__ wt, int K, int N)
{
  int idx = blockIdx.x * blockDim.x + threadIdx.x;
  if (idx < K * N) {
    int n = idx / K;
    int k = idx - n * K;
    wt[idx] = f2bf(w[(size_t)k * N + n]);
  }
}

// ---------------------------------------------------------------------------
// Kernel 3: bf16 WMMA GEMM  C[M][N] = A[M][K] * Bt[N][K]^T + bias
// 256 threads (8 waves). Block tile 128x128, per-wave 64x32 (4x2 WMMA tiles).
// K stepped by 32. A/B tiles staged in double-buffered LDS by the TDM
// (TENSOR_LOAD_TO_LDS issued by wave 0, synced with s_wait_tensorcnt+barrier).
// ---------------------------------------------------------------------------
template <bool BF16OUT>
__global__ __launch_bounds__(256) void gemm_bf16_wmma(
    const unsigned short* __restrict__ A,   // [M][K] bf16
    const unsigned short* __restrict__ Bt,  // [N][K] bf16 (transposed weights)
    const float* __restrict__ bias,         // [N]
    void* __restrict__ Cout,                // [M][N] bf16 or f32
    int M, int N, int K)
{
  __shared__ __align__(16) unsigned short As[2][128 * 32];  // 2 x 8 KB
  __shared__ __align__(16) unsigned short Bs[2][128 * 32];  // 2 x 8 KB

  const int ntiles = N >> 7;                 // 128-wide N tiles
  const int bm = blockIdx.x / ntiles;
  const int bn = blockIdx.x - bm * ntiles;

  const int tid  = threadIdx.x;
  const int wid  = tid >> 5;
  const int lane = tid & 31;
  const int lrow = lane & 15;
  const int half = lane >> 4;
  const int koff = half * 8;                 // ISA 16-bit A/B lane layout

  const int waveM = (wid & 1) * 64;          // 2 M-halves of 64 rows
  const int waveN = (wid >> 1) * 32;         // 4 N-quarters of 32 cols

  const unsigned short* Ablk = A  + (size_t)bm * 128 * K;
  const unsigned short* Bblk = Bt + (size_t)bn * 128 * K;

  const f32x8 zero = {0.f, 0.f, 0.f, 0.f, 0.f, 0.f, 0.f, 0.f};
  CFrag acc[4][2];
#pragma unroll
  for (int mi = 0; mi < 4; ++mi)
#pragma unroll
    for (int ni = 0; ni < 2; ++ni) acc[mi][ni].v = zero;

  const int nk = K >> 5;                     // K / 32 steps

  // Prologue: TDM-fill buffer 0 (wave 0 only; EXEC-independent DMA).
  if (wid == 0) {
    tdm_load_2d_bf16(lds_off(&As[0][0]), Ablk, (unsigned)K, 128u, (unsigned)K, 32u, 128u);
    tdm_load_2d_bf16(lds_off(&Bs[0][0]), Bblk, (unsigned)K, 128u, (unsigned)K, 32u, 128u);
  }

  for (int it = 0; it < nk; ++it) {
    const int buf = it & 1;
    // Prefetch next k-slice into the other buffer (readers of it were fenced
    // by the barrier at the end of the previous iteration).
    if (wid == 0 && (it + 1) < nk) {
      const int k1 = (it + 1) << 5;
      tdm_load_2d_bf16(lds_off(&As[buf ^ 1][0]), Ablk + k1, (unsigned)K, 128u,
                       (unsigned)K, 32u, 128u);
      tdm_load_2d_bf16(lds_off(&Bs[buf ^ 1][0]), Bblk + k1, (unsigned)K, 128u,
                       (unsigned)K, 32u, 128u);
    }
    // Current buffer's 2 TDM ops complete when <=2 remain outstanding (in-order).
    if ((it + 1) < nk) __builtin_amdgcn_s_wait_tensorcnt(2);
    else               __builtin_amdgcn_s_wait_tensorcnt(0);
    __syncthreads();

    AFrag a[4], b[2];
#pragma unroll
    for (int mi = 0; mi < 4; ++mi) {
      const unsigned short* ap = &As[buf][(waveM + mi * 16 + lrow) * 32 + koff];
      a[mi].q[0] = *(const u32x4*)(ap);
      a[mi].q[1] = *(const u32x4*)(ap + 16);
    }
#pragma unroll
    for (int ni = 0; ni < 2; ++ni) {
      const unsigned short* bp = &Bs[buf][(waveN + ni * 16 + lrow) * 32 + koff];
      b[ni].q[0] = *(const u32x4*)(bp);
      b[ni].q[1] = *(const u32x4*)(bp + 16);
    }
#pragma unroll
    for (int mi = 0; mi < 4; ++mi)
#pragma unroll
      for (int ni = 0; ni < 2; ++ni)
        acc[mi][ni].v = __builtin_amdgcn_wmma_f32_16x16x32_bf16(
            false, a[mi].v, false, b[ni].v, (short)0, acc[mi][ni].v, false, false);

    __syncthreads();   // all waves done reading `buf` before TDM refills it
  }

  // Epilogue: lane holds column (l&15), rows r + 8*(l>>4) of each 16x16 tile.
#pragma unroll
  for (int ni = 0; ni < 2; ++ni) {
    const int col = bn * 128 + waveN + ni * 16 + lrow;
    const float bv = bias[col];
#pragma unroll
    for (int mi = 0; mi < 4; ++mi) {
#pragma unroll
      for (int r = 0; r < 8; ++r) {
        const int row = bm * 128 + waveM + mi * 16 + r + 8 * half;
        const float v = acc[mi][ni].f[r] + bv;
        if (BF16OUT)
          ((unsigned short*)Cout)[(size_t)row * N + col] = f2bf(v);
        else
          ((float*)Cout)[(size_t)row * N + col] = v;
      }
    }
  }
}

// ---------------------------------------------------------------------------
// Kernel 4: windowed attention, one block (4 waves) per (window, head).
//   logits = (Q K^T)*scale + rel_bias + mask ; softmax ; out = P V
// ---------------------------------------------------------------------------
__global__ __launch_bounds__(128) void win_attention_kernel(
    const unsigned short* __restrict__ qkv,   // [NTOK][1536] bf16: Q|K|V
    const float* __restrict__ mask,           // [8][8][64][64]
    const float* __restrict__ bias_table,     // [225][16]
    unsigned short* __restrict__ ctx)         // [NTOK][512] bf16
{
  __shared__ __align__(16) unsigned short Qs[EWIN * HDIM];   // [64][32]
  __shared__ __align__(16) unsigned short Ks[EWIN * HDIM];   // [64][32]
  __shared__ __align__(16) unsigned short Vt[HDIM * EWIN];   // [32][64] (V^T)
  __shared__ __align__(16) unsigned short Ps[EWIN * EWIN];   // [64][64]

  const int bid  = blockIdx.x;
  const int head = bid & (NHEAD - 1);
  const int win  = bid >> 4;
  const int n    = win >> 6;
  const int wy   = (win >> 3) & 7;
  const int wx   = win & 7;

  const int tid  = threadIdx.x;
  const int wid  = tid >> 5;
  const int lane = tid & 31;
  const int lrow = lane & 15;
  const int half = lane >> 4;
  const int koff = half * 8;

  // token index for window-local row e: t = n*4096 + (wy*8 + e/8)*64 + wx*8 + e%8
  const size_t tokBase = (size_t)n * 4096 + (size_t)wy * 512 + (size_t)wx * 8;

  // Stage Q, K, V^T into LDS: 256 chunks of 8 bf16, 2 per thread
#pragma unroll
  for (int c = 0; c < 2; ++c) {
    const int idx = tid + 128 * c;
    const int e   = idx >> 2;
    const int ch  = idx & 3;
    const size_t t = tokBase + (size_t)(e >> 3) * 64 + (e & 7);
    const unsigned short* base = qkv + t * QKVC + head * HDIM + ch * 8;
    u32x4 qv = *(const u32x4*)(base);
    u32x4 kv = *(const u32x4*)(base + CDIM);
    u32x4 vv = *(const u32x4*)(base + 2 * CDIM);
    *(u32x4*)(Qs + e * HDIM + ch * 8) = qv;
    *(u32x4*)(Ks + e * HDIM + ch * 8) = kv;
    const unsigned short* vp = (const unsigned short*)&vv;
#pragma unroll
    for (int q = 0; q < 8; ++q) Vt[(ch * 8 + q) * EWIN + e] = vp[q];
  }
  __syncthreads();

  // --- logits: wave `wid` handles rows 16*wid .. 16*wid+15, all 64 cols ---
  AFrag aq;
  {
    const unsigned short* qp = Qs + (16 * wid + lrow) * HDIM + koff;
    aq.q[0] = *(const u32x4*)(qp);
    aq.q[1] = *(const u32x4*)(qp + 16);
  }
  const f32x8 zero = {0.f, 0.f, 0.f, 0.f, 0.f, 0.f, 0.f, 0.f};
  CFrag acc[4];
#pragma unroll
  for (int t = 0; t < 4; ++t) {
    AFrag bk;
    const unsigned short* kp = Ks + (t * 16 + lrow) * HDIM + koff;
    bk.q[0] = *(const u32x4*)(kp);
    bk.q[1] = *(const u32x4*)(kp + 16);
    acc[t].v = __builtin_amdgcn_wmma_f32_16x16x32_bf16(
        false, aq.v, false, bk.v, (short)0, zero, false, false);
  }

  // --- scale + rel-pos bias + mask, then row softmax (16-lane groups) ---
  const float scale = 0.17677669529663687f;  // 1/sqrt(32)
  const float* mwin = mask + ((size_t)(wy * 8 + wx)) * (EWIN * EWIN);
#pragma unroll
  for (int r = 0; r < 8; ++r) {
    const int m  = 16 * wid + r + 8 * half;   // logits row
    const int i1 = m >> 3, j1 = m & 7;
    float x[4];
#pragma unroll
    for (int t = 0; t < 4; ++t) {
      const int col = t * 16 + lrow;
      const int i2 = col >> 3, j2 = col & 7;
      const int bidx = ((i1 - i2 + 7) * 15 + (j1 - j2 + 7)) * NHEAD + head;
      x[t] = acc[t].f[r] * scale + bias_table[bidx] + mwin[m * EWIN + col];
    }
    float mx = fmaxf(fmaxf(x[0], x[1]), fmaxf(x[2], x[3]));
#pragma unroll
    for (int s = 1; s < 16; s <<= 1) mx = fmaxf(mx, __shfl_xor(mx, s, 16));
    float e0 = __expf(x[0] - mx), e1 = __expf(x[1] - mx);
    float e2 = __expf(x[2] - mx), e3 = __expf(x[3] - mx);
    float sm = e0 + e1 + e2 + e3;
#pragma unroll
    for (int s = 1; s < 16; s <<= 1) sm += __shfl_xor(sm, s, 16);
    const float inv = 1.0f / sm;
    Ps[m * EWIN +  0 + lrow] = f2bf(e0 * inv);
    Ps[m * EWIN + 16 + lrow] = f2bf(e1 * inv);
    Ps[m * EWIN + 32 + lrow] = f2bf(e2 * inv);
    Ps[m * EWIN + 48 + lrow] = f2bf(e3 * inv);
  }
  // P rows for this wave were produced by this wave; no barrier needed.

  // --- out = P(16x64) * V(64x32): 2 N-tiles, K stepped 32 ---
  CFrag outf[2];
  outf[0].v = zero; outf[1].v = zero;
#pragma unroll
  for (int ks = 0; ks < EWIN; ks += 32) {
    AFrag ap;
    const unsigned short* pp = Ps + (16 * wid + lrow) * EWIN + ks + koff;
    ap.q[0] = *(const u32x4*)(pp);
    ap.q[1] = *(const u32x4*)(pp + 16);
#pragma unroll
    for (int t = 0; t < 2; ++t) {
      AFrag bv;
      const unsigned short* vp = Vt + (t * 16 + lrow) * EWIN + ks + koff;
      bv.q[0] = *(const u32x4*)(vp);
      bv.q[1] = *(const u32x4*)(vp + 16);
      outf[t].v = __builtin_amdgcn_wmma_f32_16x16x32_bf16(
          false, ap.v, false, bv.v, (short)0, outf[t].v, false, false);
    }
  }

  // --- write context in final [token][head*32+d] layout (bf16) ---
#pragma unroll
  for (int t = 0; t < 2; ++t) {
    const int d = t * 16 + lrow;
#pragma unroll
    for (int r = 0; r < 8; ++r) {
      const int m = 16 * wid + r + 8 * half;
      const size_t tok = tokBase + (size_t)(m >> 3) * 64 + (m & 7);
      ctx[tok * CDIM + head * HDIM + d] = f2bf(outf[t].f[r]);
    }
  }
}

// ---------------------------------------------------------------------------
// Workspace layout (bytes)
// ---------------------------------------------------------------------------
#define WS_XBF   0ULL                      // 131072*512*2   = 134217728
#define WS_WQKV  134217728ULL              // 1536*512*2     =   1572864
#define WS_WOUT  135790592ULL              // 512*512*2      =    524288
#define WS_QKV   136314880ULL              // 131072*1536*2  = 402653184
#define WS_CTX   538968064ULL              // 131072*512*2   = 134217728

extern "C" void kernel_launch(void* const* d_in, const int* in_sizes, int n_in,
                              void* d_out, int out_size, void* d_ws, size_t ws_size,
                              hipStream_t stream) {
  const float* x          = (const float*)d_in[0];
  const float* mask       = (const float*)d_in[1];
  const float* qkv_w      = (const float*)d_in[2];
  const float* qkv_b      = (const float*)d_in[3];
  const float* out_w      = (const float*)d_in[4];
  const float* out_b      = (const float*)d_in[5];
  const float* bias_table = (const float*)d_in[6];
  float* out = (float*)d_out;

  char* ws = (char*)d_ws;
  unsigned short* Xbf   = (unsigned short*)(ws + WS_XBF);
  unsigned short* Wqkv  = (unsigned short*)(ws + WS_WQKV);
  unsigned short* Wout  = (unsigned short*)(ws + WS_WOUT);
  unsigned short* QKVbf = (unsigned short*)(ws + WS_QKV);
  unsigned short* CTX   = (unsigned short*)(ws + WS_CTX);

  // 1) x -> bf16 (134 MB: fits L2 for the QKV GEMM's repeated A reads)
  {
    long long n4 = (long long)NTOK * CDIM / 4;
    cvt_bf16_kernel<<<(unsigned)((n4 + 255) / 256), 256, 0, stream>>>(x, Xbf, n4);
  }
  // 2) weights -> transposed bf16
  transpose_bf16_kernel<<<(QKVC * CDIM) / 256, 256, 0, stream>>>(qkv_w, Wqkv, CDIM, QKVC);
  transpose_bf16_kernel<<<(CDIM * CDIM) / 256, 256, 0, stream>>>(out_w, Wout, CDIM, CDIM);
  // 3) QKV projection: [131072 x 512] * [512 x 1536] -> bf16
  gemm_bf16_wmma<true><<<(NTOK / 128) * (QKVC / 128), 256, 0, stream>>>(
      Xbf, Wqkv, qkv_b, (void*)QKVbf, NTOK, QKVC, CDIM);
  // 4) windowed attention: 2048 windows * 16 heads
  win_attention_kernel<<<2048 * NHEAD, 128, 0, stream>>>(QKVbf, mask, bias_table, CTX);
  // 5) output projection: [131072 x 512] * [512 x 512] -> f32 d_out
  gemm_bf16_wmma<false><<<(NTOK / 128) * (CDIM / 128), 256, 0, stream>>>(
      CTX, Wout, out_b, (void*)out, NTOK, CDIM, CDIM);
}
